// PointNetPPClassification_37718402793771
// MI455X (gfx1250) — compile-verified, matrix operand order checked by reading, not run on hardware
//
#include <hip/hip_runtime.h>
#include <hip/hip_bf16.h>

typedef __attribute__((ext_vector_type(16))) _Float16 v16h;
typedef __attribute__((ext_vector_type(8)))  float    v8f;
typedef _Float16 half_t;

// ---------------------------------------------------------------------------
// WMMA fragment loader (16x32 f16 A-tile layout, ISA 7.12.2; B uses the same
// K-striping with N in place of M). Rows are row-major with leading dim `ld`
// (halves, multiple of 32). kb is 16B aligned -> two b128 loads.
// ---------------------------------------------------------------------------
__device__ __forceinline__ v16h ld_frag(const half_t* base, int ld, int row0,
                                        int kc, int lane) {
    const int r  = row0 + (lane & 15);
    const int kb = kc + ((lane >> 4) << 3);
    const half_t* p = base + (size_t)r * ld + kb;
    union { v16h v; uint4 u[2]; } t;
    t.u[0] = *(const uint4*)(p);        // halves 0..7  -> K = kb .. kb+7
    t.u[1] = *(const uint4*)(p + 16);   // halves 8..15 -> K = kb+16 .. kb+23
    return t.v;
}

__device__ __forceinline__ v8f wmma_f16(v16h a, v16h b, v8f c) {
    return __builtin_amdgcn_wmma_f32_16x16x32_f16(false, a, false, b,
                                                  (short)0, c, false, false);
}

// ---------------------------------------------------------------------------
// Fold eval-BN into weights: W'[o,k] = W[o,k]*g[o]*rsqrt(rv[o]+eps) (f16,
// zero-padded to kpad x coutpad), b'[o] = b[o]-rm[o]*s (f32). If g==null,
// plain linear with optional bias.
// ---------------------------------------------------------------------------
__global__ void fold_kernel(const float* __restrict__ W, const float* __restrict__ g,
                            const float* __restrict__ b, const float* __restrict__ rm,
                            const float* __restrict__ rv, const float* __restrict__ plain_bias,
                            half_t* __restrict__ Wf, float* __restrict__ bf,
                            int cout, int cin, int kpad, int coutpad) {
    const int total = coutpad * kpad;
    for (int i = blockIdx.x * blockDim.x + threadIdx.x; i < total;
         i += gridDim.x * blockDim.x) {
        const int o = i / kpad, k = i % kpad;
        float s = 1.0f, bias = 0.0f;
        if (o < cout) {
            if (g) {
                s    = g[o] * rsqrtf(rv[o] + 1e-5f);
                bias = b[o] - rm[o] * s;
            } else if (plain_bias) {
                bias = plain_bias[o];
            }
        }
        const float w = (o < cout && k < cin) ? W[(size_t)o * cin + k] * s : 0.0f;
        Wf[i] = (half_t)w;
        if (k == 0) bf[o] = (o < cout) ? bias : 0.0f;
    }
}

// ---------------------------------------------------------------------------
// Exact kNN top-64: one block (256 threads) per query. Distances in LDS,
// 64 rounds of tree-reduced argmin.
// ---------------------------------------------------------------------------
template <int NSRC>
__global__ __launch_bounds__(256) void knn64_kernel(const float* __restrict__ src,
                                                    const float* __restrict__ qry,
                                                    int* __restrict__ out, int Mq) {
    __shared__ float d2[NSRC];
    __shared__ float rmin[256];
    __shared__ int   ridx[256];
    const int q = blockIdx.x;
    const int b = q / Mq, m = q % Mq;
    const float qx = qry[(size_t)(b * Mq + m) * 3 + 0];
    const float qy = qry[(size_t)(b * Mq + m) * 3 + 1];
    const float qz = qry[(size_t)(b * Mq + m) * 3 + 2];
    const float* sb = src + (size_t)b * NSRC * 3;
    for (int i = threadIdx.x; i < NSRC; i += 256) {
        const float dx = sb[i * 3 + 0] - qx;
        const float dy = sb[i * 3 + 1] - qy;
        const float dz = sb[i * 3 + 2] - qz;
        d2[i] = dx * dx + dy * dy + dz * dz;
    }
    __syncthreads();
    for (int kk = 0; kk < 64; ++kk) {
        float best = 3.4e38f;
        int   bi   = 0;
        for (int i = threadIdx.x; i < NSRC; i += 256)
            if (d2[i] < best) { best = d2[i]; bi = i; }
        rmin[threadIdx.x] = best;
        ridx[threadIdx.x] = bi;
        __syncthreads();
        for (int s = 128; s > 0; s >>= 1) {
            if (threadIdx.x < s) {
                if (rmin[threadIdx.x + s] < rmin[threadIdx.x]) {
                    rmin[threadIdx.x] = rmin[threadIdx.x + s];
                    ridx[threadIdx.x] = ridx[threadIdx.x + s];
                }
            }
            __syncthreads();
        }
        if (threadIdx.x == 0) {
            out[(size_t)q * 64 + kk] = ridx[0];
            d2[ridx[0]] = 3.4e38f;
        }
        __syncthreads();
    }
}

// ---------------------------------------------------------------------------
// One MLP stage inside a workgroup: 64 rows in LDS, wave w owns rows
// [16w,16w+16). Fused bias + ReLU epilogue, f16 result back to LDS.
// ---------------------------------------------------------------------------
template <int KP, int COUT>
__device__ __forceinline__ void mlp_stage(const half_t* x, const half_t* __restrict__ W,
                                          const float* __restrict__ bias, half_t* y,
                                          int lane, int m0) {
    for (int n0 = 0; n0 < COUT; n0 += 16) {
        v8f c = {};
#pragma unroll
        for (int kc = 0; kc < KP; kc += 32) {
            const v16h a  = ld_frag(x, KP, m0, kc, lane);
            const v16h bf = ld_frag(W, KP, n0, kc, lane);
            c = wmma_f16(a, bf, c);
        }
        const int   n  = n0 + (lane & 15);
        const float bb = bias[n];
        const int   mb = m0 + ((lane >> 4) << 3);
#pragma unroll
        for (int v = 0; v < 8; ++v) {
            const float val = fmaxf(c[v] + bb, 0.0f);
            y[(size_t)(mb + v) * COUT + n] = (half_t)val;
        }
    }
}

// ---------------------------------------------------------------------------
// Fused SA layer: gather (rel xyz || feats) for 64 kNN neighbors of one query
// into LDS, run 3-layer MLP via WMMA, max-pool over neighbors -> f16 feature.
// ---------------------------------------------------------------------------
template <int KP1, int C1, int C2, int C3, bool HAS_FEATS, int CFEAT>
__global__ __launch_bounds__(128) void sa_layer_kernel(
    const float* __restrict__ loc_src, const float* __restrict__ loc_q,
    const half_t* __restrict__ feats, const int* __restrict__ knn,
    const half_t* __restrict__ W1, const float* __restrict__ b1,
    const half_t* __restrict__ W2, const float* __restrict__ b2,
    const half_t* __restrict__ W3, const float* __restrict__ b3,
    half_t* __restrict__ out, int Nsrc, int Mq) {
    constexpr int NB = 64;
    constexpr int S0 = NB * (KP1 > C2 ? KP1 : C2);
    constexpr int S1 = NB * (C1 > C3 ? C1 : C3);
    __shared__ alignas(16) half_t buf0[S0];
    __shared__ alignas(16) half_t buf1[S1];
    __shared__ int nidx[NB];

    const int q = blockIdx.x;
    const int b = q / Mq, m = q % Mq;
    const int lane = threadIdx.x & 31;
    const int m0   = (threadIdx.x >> 5) << 4;

    if (threadIdx.x < NB) nidx[threadIdx.x] = knn[(size_t)q * NB + threadIdx.x];
    __syncthreads();

    const float qx = loc_q[(size_t)(b * Mq + m) * 3 + 0];
    const float qy = loc_q[(size_t)(b * Mq + m) * 3 + 1];
    const float qz = loc_q[(size_t)(b * Mq + m) * 3 + 2];

    for (int t = threadIdx.x; t < NB * KP1; t += 128) {
        const int j = t / KP1, c = t % KP1;
        const int id = nidx[j];
        float v = 0.0f;
        if (c < 3) {
            v = loc_src[((size_t)b * Nsrc + id) * 3 + c] -
                ((c == 0) ? qx : (c == 1) ? qy : qz);
        } else if (HAS_FEATS && c < 3 + CFEAT) {
            v = (float)feats[((size_t)b * Nsrc + id) * CFEAT + (c - 3)];
        }
        buf0[t] = (half_t)v;
    }
    __syncthreads();

    mlp_stage<KP1, C1>(buf0, W1, b1, buf1, lane, m0);
    __syncthreads();
    mlp_stage<C1, C2>(buf1, W2, b2, buf0, lane, m0);
    __syncthreads();
    mlp_stage<C2, C3>(buf0, W3, b3, buf1, lane, m0);
    __syncthreads();

    // max over 64 neighbors (post-ReLU values are >= 0)
    for (int c = threadIdx.x; c < C3; c += 128) {
        float mx = 0.0f;
        for (int j = 0; j < NB; ++j) mx = fmaxf(mx, (float)buf1[j * C3 + c]);
        out[(size_t)q * C3 + c] = (half_t)mx;
    }
}

// ---------------------------------------------------------------------------
// Flat GEMM: out = act(A @ W^T + bias). A [rows,kpad] f16 row-major,
// W [coutpad,kpad] f16. Block = 4 waves -> 64x64 output tile.
// ---------------------------------------------------------------------------
template <bool RELU, bool OUTF16>
__global__ __launch_bounds__(128) void gemm_bias_act_kernel(
    const half_t* __restrict__ A, const half_t* __restrict__ W,
    const float* __restrict__ bias, void* __restrict__ outv,
    int rows, int kpad, int ldo) {
    const int lane = threadIdx.x & 31;
    const int m0   = blockIdx.x * 64 + ((threadIdx.x >> 5) << 4);
    if (m0 + 16 > rows) return;
    const int nbase = blockIdx.y * 64;
    for (int nt = 0; nt < 4; ++nt) {
        const int n0 = nbase + nt * 16;
        v8f c = {};
        for (int kc = 0; kc < kpad; kc += 32) {
            const v16h a  = ld_frag(A, kpad, m0, kc, lane);
            const v16h bf = ld_frag(W, kpad, n0, kc, lane);
            c = wmma_f16(a, bf, c);
        }
        const int   n  = n0 + (lane & 15);
        const float bb = bias[n];
        const int   mb = m0 + ((lane >> 4) << 3);
#pragma unroll
        for (int v = 0; v < 8; ++v) {
            float val = c[v] + bb;
            if (RELU) val = fmaxf(val, 0.0f);
            if (OUTF16)
                ((half_t*)outv)[(size_t)(mb + v) * ldo + n] = (half_t)val;
            else
                ((float*)outv)[(size_t)(mb + v) * ldo + n] = val;
        }
    }
}

// Build final-stage input: A0[b*256+m, 0:3]=loc2, [3:259]=f2, rest 0.
__global__ void prep_final_kernel(const float* __restrict__ loc2,
                                  const half_t* __restrict__ f2,
                                  half_t* __restrict__ A0) {
    const int r = blockIdx.x;               // 0..4095
    for (int c = threadIdx.x; c < 288; c += blockDim.x) {
        float v = 0.0f;
        if (c < 3)        v = loc2[(size_t)r * 3 + c];
        else if (c < 259) v = (float)f2[(size_t)r * 256 + (c - 3)];
        A0[(size_t)r * 288 + c] = (half_t)v;
    }
}

// Per-batch column max over rowsPerGroup rows (post-ReLU, so init 0).
__global__ void colmax_kernel(const half_t* __restrict__ A, half_t* __restrict__ out,
                              int rowsPerGroup, int cols) {
    const int b = blockIdx.y;
    const int c = blockIdx.x * blockDim.x + threadIdx.x;
    if (c >= cols) return;
    const half_t* base = A + (size_t)b * rowsPerGroup * cols;
    float mx = 0.0f;
    for (int j = 0; j < rowsPerGroup; ++j)
        mx = fmaxf(mx, (float)base[(size_t)j * cols + c]);
    out[(size_t)b * cols + c] = (half_t)mx;
}

__global__ void extract_out_kernel(const float* __restrict__ t, float* __restrict__ out) {
    const int i = blockIdx.x * blockDim.x + threadIdx.x;   // 16*40
    if (i < 16 * 40) out[i] = t[(i / 40) * 64 + (i % 40)];
}

// ---------------------------------------------------------------------------
// Host-side orchestration
// ---------------------------------------------------------------------------
extern "C" void kernel_launch(void* const* d_in, const int* in_sizes, int n_in,
                              void* d_out, int out_size, void* d_ws, size_t ws_size,
                              hipStream_t stream) {
    (void)in_sizes; (void)n_in; (void)out_size; (void)ws_size;

    const float* loc0 = (const float*)d_in[0];   // [16,4096,3]
    const float* loc1 = (const float*)d_in[1];   // [16,1024,3]
    const float* loc2 = (const float*)d_in[2];   // [16,256,3]
    // p_sa1 at 3..17, p_sa2 at 18..32, p_fin at 33..47 (each layer: W,g,b,rm,rv)
    const int SA1 = 3, SA2 = 18, FIN = 33;
    const float* fcW1 = (const float*)d_in[48];
    const float* bn1g = (const float*)d_in[49];
    const float* bn1b = (const float*)d_in[50];
    const float* bn1m = (const float*)d_in[51];
    const float* bn1v = (const float*)d_in[52];
    const float* fcW2 = (const float*)d_in[53];
    const float* bn2g = (const float*)d_in[54];
    const float* bn2b = (const float*)d_in[55];
    const float* bn2m = (const float*)d_in[56];
    const float* bn2v = (const float*)d_in[57];
    const float* fcW3 = (const float*)d_in[58];
    const float* fcb3 = (const float*)d_in[59];

    size_t off = 0;
    auto alloc = [&](size_t bytes) -> void* {
        void* p = (char*)d_ws + off;
        off += (bytes + 255) & ~(size_t)255;
        return p;
    };

    // folded weights (f16) + biases (f32)
    half_t* sa1W1 = (half_t*)alloc(64  * 32  * 2); float* sa1b1 = (float*)alloc(64  * 4);
    half_t* sa1W2 = (half_t*)alloc(64  * 64  * 2); float* sa1b2 = (float*)alloc(64  * 4);
    half_t* sa1W3 = (half_t*)alloc(128 * 64  * 2); float* sa1b3 = (float*)alloc(128 * 4);
    half_t* sa2W1 = (half_t*)alloc(128 * 160 * 2); float* sa2b1 = (float*)alloc(128 * 4);
    half_t* sa2W2 = (half_t*)alloc(128 * 128 * 2); float* sa2b2 = (float*)alloc(128 * 4);
    half_t* sa2W3 = (half_t*)alloc(256 * 128 * 2); float* sa2b3 = (float*)alloc(256 * 4);
    half_t* finW1 = (half_t*)alloc(256 * 288 * 2); float* finb1 = (float*)alloc(256 * 4);
    half_t* finW2 = (half_t*)alloc(512 * 256 * 2); float* finb2 = (float*)alloc(512 * 4);
    half_t* finW3 = (half_t*)alloc(1024 * 512 * 2); float* finb3 = (float*)alloc(1024 * 4);
    half_t* fcW1f = (half_t*)alloc(512 * 1024 * 2); float* fcb1f = (float*)alloc(512 * 4);
    half_t* fcW2f = (half_t*)alloc(256 * 512 * 2);  float* fcb2f = (float*)alloc(256 * 4);
    half_t* fcW3f = (half_t*)alloc(64  * 256 * 2);  float* fcb3f = (float*)alloc(64  * 4);
    // intermediates
    int*    knn1 = (int*)alloc((size_t)16384 * 64 * 4);
    int*    knn2 = (int*)alloc((size_t)4096 * 64 * 4);
    half_t* f1   = (half_t*)alloc((size_t)16 * 1024 * 128 * 2);
    half_t* f2   = (half_t*)alloc((size_t)16 * 256 * 256 * 2);
    half_t* A0   = (half_t*)alloc((size_t)4096 * 288 * 2);
    half_t* A1   = (half_t*)alloc((size_t)4096 * 256 * 2);
    half_t* A2   = (half_t*)alloc((size_t)4096 * 512 * 2);
    half_t* A3   = (half_t*)alloc((size_t)4096 * 1024 * 2);
    half_t* f3   = (half_t*)alloc((size_t)16 * 1024 * 2);
    half_t* t1   = (half_t*)alloc((size_t)16 * 512 * 2);
    half_t* t2   = (half_t*)alloc((size_t)16 * 256 * 2);
    float*  t3   = (float*)alloc((size_t)16 * 64 * 4);

    auto fold_bn = [&](int base, int layer, half_t* Wf, float* bf,
                       int cout, int cin, int kpad, int coutpad) {
        const int i = base + 5 * layer;
        const int total  = coutpad * kpad;
        const int blocks = (total + 255) / 256;
        fold_kernel<<<blocks, 256, 0, stream>>>(
            (const float*)d_in[i + 0], (const float*)d_in[i + 1],
            (const float*)d_in[i + 2], (const float*)d_in[i + 3],
            (const float*)d_in[i + 4], nullptr, Wf, bf, cout, cin, kpad, coutpad);
    };
    fold_bn(SA1, 0, sa1W1, sa1b1, 64, 3, 32, 64);
    fold_bn(SA1, 1, sa1W2, sa1b2, 64, 64, 64, 64);
    fold_bn(SA1, 2, sa1W3, sa1b3, 128, 64, 64, 128);
    fold_bn(SA2, 0, sa2W1, sa2b1, 128, 131, 160, 128);
    fold_bn(SA2, 1, sa2W2, sa2b2, 128, 128, 128, 128);
    fold_bn(SA2, 2, sa2W3, sa2b3, 256, 128, 128, 256);
    fold_bn(FIN, 0, finW1, finb1, 256, 259, 288, 256);
    fold_bn(FIN, 1, finW2, finb2, 512, 256, 256, 512);
    fold_bn(FIN, 2, finW3, finb3, 1024, 512, 512, 1024);
    fold_kernel<<<(512 * 1024 + 255) / 256, 256, 0, stream>>>(
        fcW1, bn1g, bn1b, bn1m, bn1v, nullptr, fcW1f, fcb1f, 512, 1024, 1024, 512);
    fold_kernel<<<(256 * 512 + 255) / 256, 256, 0, stream>>>(
        fcW2, bn2g, bn2b, bn2m, bn2v, nullptr, fcW2f, fcb2f, 256, 512, 512, 256);
    fold_kernel<<<(64 * 256 + 255) / 256, 256, 0, stream>>>(
        fcW3, nullptr, nullptr, nullptr, nullptr, fcb3, fcW3f, fcb3f, 40, 256, 256, 64);

    // kNN
    knn64_kernel<4096><<<16384, 256, 0, stream>>>(loc0, loc1, knn1, 1024);
    knn64_kernel<1024><<<4096, 256, 0, stream>>>(loc1, loc2, knn2, 256);

    // SA layers (fused gather + 3-layer WMMA MLP + neighbor max)
    sa_layer_kernel<32, 64, 64, 128, false, 0><<<16384, 128, 0, stream>>>(
        loc0, loc1, nullptr, knn1, sa1W1, sa1b1, sa1W2, sa1b2, sa1W3, sa1b3,
        f1, 4096, 1024);
    sa_layer_kernel<160, 128, 128, 256, true, 128><<<4096, 128, 0, stream>>>(
        loc1, loc2, f1, knn2, sa2W1, sa2b1, sa2W2, sa2b2, sa2W3, sa2b3,
        f2, 1024, 256);

    // Final PointNet stage: flat GEMM chain over [4096, *], then per-batch max
    prep_final_kernel<<<4096, 128, 0, stream>>>(loc2, f2, A0);
    gemm_bias_act_kernel<true, true><<<dim3(64, 4), 128, 0, stream>>>(
        A0, finW1, finb1, A1, 4096, 288, 256);
    gemm_bias_act_kernel<true, true><<<dim3(64, 8), 128, 0, stream>>>(
        A1, finW2, finb2, A2, 4096, 256, 512);
    gemm_bias_act_kernel<true, true><<<dim3(64, 16), 128, 0, stream>>>(
        A2, finW3, finb3, A3, 4096, 512, 1024);
    colmax_kernel<<<dim3(8, 16), 128, 0, stream>>>(A3, f3, 256, 1024);

    // FC head
    gemm_bias_act_kernel<true, true><<<dim3(1, 8), 128, 0, stream>>>(
        f3, fcW1f, fcb1f, t1, 16, 1024, 512);
    gemm_bias_act_kernel<true, true><<<dim3(1, 4), 128, 0, stream>>>(
        t1, fcW2f, fcb2f, t2, 16, 512, 256);
    gemm_bias_act_kernel<false, false><<<dim3(1, 1), 128, 0, stream>>>(
        t2, fcW3f, fcb3f, t3, 16, 256, 64);
    extract_out_kernel<<<(16 * 40 + 127) / 128, 128, 0, stream>>>(t3, (float*)d_out);
}
